// Model_17325898072228
// MI455X (gfx1250) — compile-verified, matrix operand order checked by reading, not run on hardware
//
#include <hip/hip_runtime.h>
#include <hip/hip_bf16.h>

typedef _Float16 f16;
typedef __attribute__((ext_vector_type(16))) _Float16 v16h;
typedef __attribute__((ext_vector_type(8)))  float    v8f;

#define B_ 32
#define N_ 1024
#define I_ 128
#define L_ 32
#define C_ 16
#define LDP 136   // padded LDS row stride in halfs (128 + 8) -> 272B, bank-conflict free, 16B aligned

// Load a 16x32 f16 WMMA operand tile from LDS.
// base points at element [row][0] of a padded (stride LDP) f16 matrix.
// Lane half hi=0 takes K = k0+0..7 and k0+16..23; hi=1 takes K = k0+8..15 and k0+24..31.
__device__ __forceinline__ v16h load_tile(const f16* __restrict__ base, int k0, int ksel) {
  union { v16h v; uint4 u[2]; } t;
  t.u[0] = *(const uint4*)(base + k0 + ksel);
  t.u[1] = *(const uint4*)(base + k0 + 16 + ksel);
  return t.v;
}

__global__ __launch_bounds__(256) void mil_attn_kernel(
    const float* __restrict__ bags, const float* __restrict__ loc,
    const float* __restrict__ W1,   const float* __restrict__ b1,
    const float* __restrict__ Wa,   const float* __restrict__ ba,
    const float* __restrict__ Wp,   const float* __restrict__ bp,
    float* __restrict__ out)
{
  extern __shared__ char smem[];
  f16*   w1t  = (f16*)smem;                        // [I_][LDP] transposed W1[c]: w1t[j*LDP + i]
  f16*   bt   = (f16*)(smem + I_ * LDP * 2);       // [128][LDP] bags chunk (f16)
  float* b1s  = (float*)(smem + 2 * I_ * LDP * 2); // [I_]
  float* was  = b1s + I_;                          // [I_]
  float* wm   = was + I_;                          // [8]
  float* wl   = wm + 8;                            // [8]
  float* wacc = wl + 8;                            // [8][I_]

  const int c    = blockIdx.x;
  const int b    = blockIdx.y;
  const int tid  = threadIdx.x;
  const int wv   = tid >> 5;        // wave id 0..7
  const int lane = tid & 31;
  const int l16  = lane & 15;
  const int hi   = lane >> 4;       // 0 or 1 (lane half)
  const int ksel = hi * 8;

  // ---- one-time: stage W1[c]^T (f16), b1[c], Wa[c] into LDS ----
  const float* W1c = W1 + (size_t)c * I_ * I_;
  for (int f = tid; f < I_ * I_; f += 256) {
    int i = f >> 7, j = f & 127;
    w1t[j * LDP + i] = (f16)W1c[f];               // coalesced global read, transposed LDS write
  }
  if (tid < I_) { b1s[tid] = b1[c * I_ + tid]; was[tid] = Wa[c * I_ + tid]; }
  __syncthreads();

  const float bac = ba[c];

  // hoist per-(lane,jt) loop-invariant scalars out of the N loop
  float bjv[8], wajv[8];
  #pragma unroll
  for (int jt = 0; jt < 8; ++jt) {
    bjv[jt]  = b1s[jt * 16 + l16];
    wajv[jt] = was[jt * 16 + l16];
  }

  // ---- online-softmax running state (per wave; lane-uniform after reductions) ----
  float m_run = -1e30f;
  float l_run = 0.f;
  float acc[8];                                    // weighted h accumulator, one column per jt per lane
  #pragma unroll
  for (int jt = 0; jt < 8; ++jt) acc[jt] = 0.f;

  for (int it = 0; it < N_ / 128; ++it) {
    // stage 128 rows of bags -> f16 LDS.
    // Each thread converts 8 contiguous floats (2x float4) and writes one
    // 16B ds_store_b128. 2048 float8 groups / 256 threads = 8 each.
    const float4* src = (const float4*)(bags + ((size_t)b * N_ + it * 128) * I_);
    #pragma unroll
    for (int q = 0; q < 8; ++q) {
      int fidx = tid + 256 * q;                    // float8 index, 2048 total
      float4 va = src[fidx * 2];
      float4 vb = src[fidx * 2 + 1];
      int row = fidx >> 4, col8 = (fidx & 15) * 8;
      union { f16 h[8]; uint4 u; } pk;
      pk.h[0] = (f16)va.x; pk.h[1] = (f16)va.y; pk.h[2] = (f16)va.z; pk.h[3] = (f16)va.w;
      pk.h[4] = (f16)vb.x; pk.h[5] = (f16)vb.y; pk.h[6] = (f16)vb.z; pk.h[7] = (f16)vb.w;
      *(uint4*)(&bt[row * LDP + col8]) = pk.u;     // single ds_store_b128 (16B aligned)
    }
    __syncthreads();

    // ---- this wave's 16x128 h tile via WMMA (C starts as inline 0; bias in epilogue) ----
    v8f   h[8];
    float sPart[8];
    #pragma unroll
    for (int v = 0; v < 8; ++v) sPart[v] = 0.f;

    const f16* arow = &bt[(wv * 16 + l16) * LDP];
    #pragma unroll
    for (int jt = 0; jt < 8; ++jt) {
      const f16* brow = &w1t[(jt * 16 + l16) * LDP];
      v8f cacc = {};                               // folds into WMMA SRC2 inline 0
      #pragma unroll
      for (int k0 = 0; k0 < I_; k0 += 32) {
        v16h a  = load_tile(arow, k0, ksel);
        v16h bb = load_tile(brow, k0, ksel);
        cacc = __builtin_amdgcn_wmma_f32_16x16x32_f16(
            false, a, false, bb, (short)0, cacc, false, false);
      }
      const float bj  = bjv[jt];
      const float waj = wajv[jt];
      #pragma unroll
      for (int e = 0; e < 8; ++e) {
        float hv = fmaxf(cacc[e] + bj, 0.f);       // bias + relu
        cacc[e] = hv;
        sPart[e] += hv * waj;                      // partial of s over this lane's column
      }
      h[jt] = cacc;
    }

    // ---- attention scores for the 16 rows: reduce over 16-lane column groups ----
    float s[8];
    #pragma unroll
    for (int v = 0; v < 8; ++v) {
      float x = sPart[v];
      x += __shfl_xor(x, 1, 32);
      x += __shfl_xor(x, 2, 32);
      x += __shfl_xor(x, 4, 32);
      x += __shfl_xor(x, 8, 32);                   // lane now holds full s for row (v + 8*hi)
      x += bac;
      s[v] = x > 0.f ? x : 0.f;                    // relu on scores
    }

    // ---- online softmax update ----
    float mloc = s[0];
    #pragma unroll
    for (int v = 1; v < 8; ++v) mloc = fmaxf(mloc, s[v]);
    mloc = fmaxf(mloc, __shfl_xor(mloc, 16, 32));  // max over all 16 rows of chunk
    const float m_new = fmaxf(m_run, mloc);
    const float scale = __expf(m_run - m_new);

    float e_w[8], lsum = 0.f;
    #pragma unroll
    for (int v = 0; v < 8; ++v) { e_w[v] = __expf(s[v] - m_new); lsum += e_w[v]; }
    lsum += __shfl_xor(lsum, 16, 32);
    l_run = l_run * scale + lsum;
    m_run = m_new;

    #pragma unroll
    for (int jt = 0; jt < 8; ++jt) {
      float a = 0.f;
      #pragma unroll
      for (int e = 0; e < 8; ++e) a += e_w[e] * h[jt][e];  // rows weighted by softmax numerator
      acc[jt] = acc[jt] * scale + a;
    }
    __syncthreads();                               // bt consumed; safe to overwrite next iter
  }

  // ---- merge the 8 waves' partial softmax states via LDS ----
  #pragma unroll
  for (int jt = 0; jt < 8; ++jt) {
    float a2 = acc[jt] + __shfl_xor(acc[jt], 16, 32);   // fold the two lane halves (same column)
    if (hi == 0) wacc[wv * I_ + jt * 16 + l16] = a2;
  }
  if (lane == 0) { wm[wv] = m_run; wl[wv] = l_run; }
  __syncthreads();

  if (wv == 0) {
    float M = wm[0];
    #pragma unroll
    for (int w = 1; w < 8; ++w) M = fmaxf(M, wm[w]);
    float sc[8], Ls = 0.f;
    #pragma unroll
    for (int w = 0; w < 8; ++w) { sc[w] = __expf(wm[w] - M); Ls += wl[w] * sc[w]; }
    const float inv = 1.f / Ls;

    float part = 0.f;
    #pragma unroll
    for (int q = 0; q < 4; ++q) {
      const int col = lane + 32 * q;
      float p = 0.f;
      #pragma unroll
      for (int w = 0; w < 8; ++w) p += wacc[w * I_ + col] * sc[w];
      part += (p * inv) * Wp[col];                 // pooled[col] * Wp[col]
    }
    part += loc[b * L_ + lane] * Wp[I_ + lane];    // loc part of concat feature
    part += __shfl_xor(part, 16, 32);
    part += __shfl_xor(part, 8, 32);
    part += __shfl_xor(part, 4, 32);
    part += __shfl_xor(part, 2, 32);
    part += __shfl_xor(part, 1, 32);
    if (lane == 0) {
      float y = part + bp[0];
      out[b * C_ + c] = 1.f / (1.f + __expf(-y));
    }
  }
}

extern "C" void kernel_launch(void* const* d_in, const int* in_sizes, int n_in,
                              void* d_out, int out_size, void* d_ws, size_t ws_size,
                              hipStream_t stream) {
  (void)in_sizes; (void)n_in; (void)d_ws; (void)ws_size; (void)out_size;
  const float* bags = (const float*)d_in[0];
  const float* loc  = (const float*)d_in[1];
  const float* W1   = (const float*)d_in[2];
  const float* b1   = (const float*)d_in[3];
  const float* Wa   = (const float*)d_in[4];
  const float* ba   = (const float*)d_in[5];
  const float* Wp   = (const float*)d_in[6];
  const float* bp   = (const float*)d_in[7];

  // dynamic LDS: 2 f16 matrices [128][136] + b1/Wa + 8-wave merge scratch
  size_t lds = (size_t)2 * I_ * LDP * 2      // w1t + bt
             + 2 * I_ * 4                    // b1s + was
             + 16 * 4                        // wm + wl
             + 8 * I_ * 4;                   // wacc
  dim3 grid(C_, B_);
  mil_attn_kernel<<<grid, dim3(256), lds, stream>>>(bags, loc, W1, b1, Wa, ba, Wp, bp,
                                                    (float*)d_out);
}
